// GPT2NonResidualAttention_18210661335276
// MI455X (gfx1250) — compile-verified, matrix operand order checked by reading, not run on hardware
//
#include <hip/hip_runtime.h>
#include <hip/hip_bf16.h>

// ---------------------------------------------------------------------------
// GPT2 non-residual attention for MI455X (gfx1250, wave32, WMMA + TDM)
//   B=4, S=1024, E=1024, H=16, DH=64, P=64
// Pipeline:
//   1) qkv_gemm_kernel : hidden @ c_attn_w + b  -> q/k/v  (bf16, split heads)
//   2) attn_kernel     : flash-style attention over [prompt | causal | self]
//   3) proj_gemm_kernel: attn_out @ c_proj_w + b -> fp32 output
//      (A tile staged by the Tensor Data Mover with descriptor-driven LDS pad)
// All staging loops are split into load-phase / store-phase so the compiler
// issues all global loads back-to-back (partial s_wait_loadcnt) instead of
// serializing load->convert->ds_store per element.
// ---------------------------------------------------------------------------

typedef __attribute__((ext_vector_type(16))) __bf16 v16bf;
typedef __attribute__((ext_vector_type(8)))  __bf16 v8bf;
typedef __attribute__((ext_vector_type(8)))  float  v8f;
typedef __attribute__((ext_vector_type(4)))  unsigned int u32x4;
typedef __attribute__((ext_vector_type(8)))  int i32x8;
typedef __attribute__((ext_vector_type(4)))  int i32x4;

#ifndef __has_builtin
#define __has_builtin(x) 0
#endif
#if __has_builtin(__builtin_amdgcn_tensor_load_to_lds) && \
    __has_builtin(__builtin_amdgcn_s_wait_tensorcnt)
#define USE_TDM 1
#else
#define USE_TDM 0
#endif

#define B_   4
#define S_   1024
#define E_   1024
#define H_   16
#define DH_  64
#define P_   64
#define NEG_BIAS (-10000.0f)

__device__ __forceinline__ v8f wmma_bf16(v16bf a, v16bf b, v8f c) {
  // emits v_wmma_f32_16x16x32_bf16
  return __builtin_amdgcn_wmma_f32_16x16x32_bf16(false, a, false, b,
                                                 (short)0, c, false, false);
}

// Load a 16x32 bf16 A/B fragment from a row-major tile.
// rowPtr must point at element (row = lane&15, k = 0) of the fragment window.
// ISA layout: lanes 0-15 hold K 0..7 / 16..23, lanes 16-31 hold K 8..15 / 24..31.
__device__ __forceinline__ v16bf load_frag(const __bf16* rowPtr) {
  const int lane = threadIdx.x & 31;
  const int kb = (lane < 16) ? 0 : 8;
  v8bf lo = *(const v8bf*)(rowPtr + kb);
  v8bf hi = *(const v8bf*)(rowPtr + 16 + kb);
  v16bf f;
#pragma unroll
  for (int i = 0; i < 8; ++i) { f[i] = lo[i]; f[i + 8] = hi[i]; }
  return f;
}

// Reductions across a 16-lane half-wave (rows of the C-matrix layout live in
// 16 contiguous lanes on wave32).
__device__ __forceinline__ float half_max(float v) {
  v = fmaxf(v, __shfl_xor(v, 1));
  v = fmaxf(v, __shfl_xor(v, 2));
  v = fmaxf(v, __shfl_xor(v, 4));
  v = fmaxf(v, __shfl_xor(v, 8));
  return v;
}
__device__ __forceinline__ float half_sum(float v) {
  v += __shfl_xor(v, 1);
  v += __shfl_xor(v, 2);
  v += __shfl_xor(v, 4);
  v += __shfl_xor(v, 8);
  return v;
}

// ---------------------------------------------------------------------------
// Kernel 1: QKV projection.  C[4096x3072] = A[4096x1024] @ W[1024x3072] + b
// Block 256 threads (8 waves), tile 128x128x32; wave = 32 rows x 64 cols.
// Output scattered to bf16 q/k/v in (B,H,S,DH) layout.
// ---------------------------------------------------------------------------
__global__ __launch_bounds__(256) void qkv_gemm_kernel(
    const float* __restrict__ A, const float* __restrict__ W,
    const float* __restrict__ bias,
    unsigned short* __restrict__ qb_, unsigned short* __restrict__ kb_,
    unsigned short* __restrict__ vb_) {
  __bf16* qb = reinterpret_cast<__bf16*>(qb_);
  __bf16* kb = reinterpret_cast<__bf16*>(kb_);
  __bf16* vb = reinterpret_cast<__bf16*>(vb_);

  __shared__ __align__(16) __bf16 As[128][40];  // [m][k], padded
  __shared__ __align__(16) __bf16 Bs[128][40];  // [n][k] (W transposed)

  const int tid  = threadIdx.x;
  const int wave = tid >> 5;
  const int lane = tid & 31;
  const int lh   = lane & 15;
  const int hif  = lane >> 4;
  const int wm   = wave & 3;   // 4 wave-rows of 32
  const int wn   = wave >> 2;  // 2 wave-cols of 64
  const int gm0  = blockIdx.x * 128;
  const int gn0  = blockIdx.y * 128;

  v8f acc[2][4] = {};

  for (int k0 = 0; k0 < E_; k0 += 32) {
    // ---- load phase: issue all 8 global_load_b128 before any LDS store ----
    float4 areg[4], wreg[4];
#pragma unroll
    for (int i = 0; i < 4; ++i) {
      int f = tid + i * 256;
      int r = f >> 3;
      int c = (f & 7) << 2;
      areg[i] = *(const float4*)(A + (size_t)(gm0 + r) * E_ + k0 + c);
    }
#pragma unroll
    for (int i = 0; i < 4; ++i) {
      int f  = tid + i * 256;
      int kr = f >> 5;
      int c  = (f & 31) << 2;
      wreg[i] = *(const float4*)(W + (size_t)(k0 + kr) * (3 * E_) + gn0 + c);
    }
    if (k0 + 32 < E_) {                      // next-tile prefetch into GL2
      int r = tid >> 3, c = (tid & 7) << 2;
      __builtin_prefetch(A + (size_t)(gm0 + r) * E_ + k0 + 32 + c, 0, 3);
      int kr = tid >> 5, c2 = (tid & 31) << 2;
      __builtin_prefetch(W + (size_t)(k0 + 32 + kr) * (3 * E_) + gn0 + c2, 0, 3);
    }
    // ---- store phase: convert fp32 -> bf16 and write LDS tiles ----
#pragma unroll
    for (int i = 0; i < 4; ++i) {
      int f = tid + i * 256;
      int r = f >> 3;
      int c = (f & 7) << 2;
      As[r][c + 0] = (__bf16)areg[i].x;
      As[r][c + 1] = (__bf16)areg[i].y;
      As[r][c + 2] = (__bf16)areg[i].z;
      As[r][c + 3] = (__bf16)areg[i].w;
    }
#pragma unroll
    for (int i = 0; i < 4; ++i) {
      int f  = tid + i * 256;
      int kr = f >> 5;
      int c  = (f & 31) << 2;
      Bs[c + 0][kr] = (__bf16)wreg[i].x;
      Bs[c + 1][kr] = (__bf16)wreg[i].y;
      Bs[c + 2][kr] = (__bf16)wreg[i].z;
      Bs[c + 3][kr] = (__bf16)wreg[i].w;
    }
    __syncthreads();

    v16bf af[2], bfg[4];
#pragma unroll
    for (int mi = 0; mi < 2; ++mi)
      af[mi] = load_frag(&As[wm * 32 + mi * 16 + lh][0]);
#pragma unroll
    for (int ni = 0; ni < 4; ++ni)
      bfg[ni] = load_frag(&Bs[wn * 64 + ni * 16 + lh][0]);
#pragma unroll
    for (int mi = 0; mi < 2; ++mi)
#pragma unroll
      for (int ni = 0; ni < 4; ++ni)
        acc[mi][ni] = wmma_bf16(af[mi], bfg[ni], acc[mi][ni]);
    __syncthreads();
  }

#pragma unroll
  for (int mi = 0; mi < 2; ++mi)
#pragma unroll
    for (int ni = 0; ni < 4; ++ni)
#pragma unroll
      for (int i = 0; i < 8; ++i) {
        int r = gm0 + wm * 32 + mi * 16 + i + 8 * hif;
        int c = gn0 + wn * 64 + ni * 16 + lh;
        float v = acc[mi][ni][i] + bias[c];
        int which = c >> 10;      // 0=q,1=k,2=v
        int e  = c & 1023;
        int h  = e >> 6;
        int dh = e & 63;
        int bb = r >> 10;
        int s  = r & 1023;
        size_t idx = ((((size_t)bb * H_ + h) * S_) + s) * DH_ + dh;
        __bf16 bv = (__bf16)v;
        if (which == 0) qb[idx] = bv;
        else if (which == 1) kb[idx] = bv;
        else vb[idx] = bv;
      }
}

// ---------------------------------------------------------------------------
// Kernel 2: attention.  One block (4 waves, 128 thr) per 64 query rows of a
// (b,h). Flash-style online softmax over unified keys [prompt(64) | textual].
// Self token handled as a final extra online-softmax step.
// ---------------------------------------------------------------------------
__global__ __launch_bounds__(128) void attn_kernel(
    const float* __restrict__ pK, const float* __restrict__ pV,
    const float* __restrict__ tK, const float* __restrict__ tV,
    const unsigned char* __restrict__ pMask,
    const unsigned short* __restrict__ qb_, const unsigned short* __restrict__ kb_,
    const unsigned short* __restrict__ vb_, unsigned short* __restrict__ ob_) {
  const __bf16* qb = reinterpret_cast<const __bf16*>(qb_);
  const __bf16* kb = reinterpret_cast<const __bf16*>(kb_);
  const __bf16* vb = reinterpret_cast<const __bf16*>(vb_);
  __bf16* ob = reinterpret_cast<__bf16*>(ob_);

  __shared__ __align__(16) __bf16 Kt[32][72];      // [key][dh]
  __shared__ __align__(16) __bf16 Vt[64][40];      // [dh][key] (transposed)
  __shared__ __align__(16) __bf16 Pb[4][16 * 40];  // per-wave P tile [row][key]
  __shared__ float Sself[4][16];

  const int tid  = threadIdx.x;
  const int wave = tid >> 5;
  const int lane = tid & 31;
  const int lh   = lane & 15;
  const int hif  = lane >> 4;

  const int nqb = S_ / 64;
  const int qbk = blockIdx.x % nqb;
  const int bh  = blockIdx.x / nqb;
  const int b   = bh >> 4;
  const int h   = bh & 15;
  const int qs  = qbk * 64;
  const int row0 = qs + wave * 16;

  const size_t bhS = (size_t)bh * S_;
  const size_t bhP = (size_t)bh * P_;

  // Q fragments (16 rows x 64 dh) held in registers the whole kernel.
  const __bf16* qrow = qb + (bhS + row0 + lh) * DH_;
  v16bf qa0 = load_frag(qrow);
  v16bf qa1 = load_frag(qrow + 32);

  v8f acc[4] = {};
  float M[8], L[8];
#pragma unroll
  for (int i = 0; i < 8; ++i) { M[i] = -3.0e38f; L[i] = 0.0f; }

  const int nk = P_ + qs + 64;  // keys with nonzero causal contribution
  for (int kc = 0; kc < nk; kc += 32) {
    // ---- load phase: all 8 global_load_b128 for K and V chunks ----
    float4 kreg[4], vreg[4];
#pragma unroll
    for (int i = 0; i < 4; ++i) {
      int f    = tid + i * 128;    // float4 index 0..511
      int key  = f >> 4;           // 0..31
      int d0   = (f & 15) << 2;    // 0..60
      int kidx = kc + key;
      const float* ks;
      const float* vs;
      if (kidx < P_) {
        ks = pK + (bhP + kidx) * DH_ + d0;
        vs = pV + (bhP + kidx) * DH_ + d0;
      } else {
        ks = tK + (bhS + (kidx - P_)) * DH_ + d0;
        vs = tV + (bhS + (kidx - P_)) * DH_ + d0;
      }
      kreg[i] = *(const float4*)ks;
      vreg[i] = *(const float4*)vs;
    }
    if (kc + 32 < nk) {            // prefetch next key chunk into GL2
      int key  = tid >> 2;         // 0..31
      int d0   = (tid & 3) << 4;   // 0,16,32,48
      int kidx = kc + 32 + key;
      const float* ks = (kidx < P_) ? pK + (bhP + kidx) * DH_ + d0
                                    : tK + (bhS + (kidx - P_)) * DH_ + d0;
      const float* vs = (kidx < P_) ? pV + (bhP + kidx) * DH_ + d0
                                    : tV + (bhS + (kidx - P_)) * DH_ + d0;
      __builtin_prefetch(ks, 0, 3);
      __builtin_prefetch(vs, 0, 3);
    }
    // ---- store phase: K row-major, V transposed, both bf16 ----
#pragma unroll
    for (int i = 0; i < 4; ++i) {
      int f   = tid + i * 128;
      int key = f >> 4;
      int d0  = (f & 15) << 2;
      Kt[key][d0 + 0] = (__bf16)kreg[i].x;
      Kt[key][d0 + 1] = (__bf16)kreg[i].y;
      Kt[key][d0 + 2] = (__bf16)kreg[i].z;
      Kt[key][d0 + 3] = (__bf16)kreg[i].w;
      Vt[d0 + 0][key] = (__bf16)vreg[i].x;
      Vt[d0 + 1][key] = (__bf16)vreg[i].y;
      Vt[d0 + 2][key] = (__bf16)vreg[i].z;
      Vt[d0 + 3][key] = (__bf16)vreg[i].w;
    }
    __syncthreads();

    // Scores: 16 rows x 32 keys = two 16x16 C tiles, K-dim = dh (2x32).
    v8f s0 = {};
    v8f s1 = {};
    s0 = wmma_bf16(qa0, load_frag(&Kt[lh][0]),  s0);
    s0 = wmma_bf16(qa1, load_frag(&Kt[lh][32]), s0);
    s1 = wmma_bf16(qa0, load_frag(&Kt[16 + lh][0]),  s1);
    s1 = wmma_bf16(qa1, load_frag(&Kt[16 + lh][32]), s1);

    const int k0i = kc + lh;        // key col of s0 for this lane
    const int k1i = kc + 16 + lh;   // key col of s1

    float alpha[8];
#pragma unroll
    for (int i = 0; i < 8; ++i) {
      int srow = row0 + i + 8 * hif;
      float v0 = s0[i] * 0.125f;   // 1/sqrt(64)
      float v1 = s1[i] * 0.125f;
      if (k0i < P_) {
        if (!pMask[((size_t)b * S_ + srow) * P_ + k0i]) v0 = NEG_BIAS;
      } else if (!((k0i - P_) < srow)) {
        v0 = NEG_BIAS;
      }
      if (k1i < P_) {
        if (!pMask[((size_t)b * S_ + srow) * P_ + k1i]) v1 = NEG_BIAS;
      } else if (!((k1i - P_) < srow)) {
        v1 = NEG_BIAS;
      }
      float mx = half_max(fmaxf(v0, v1));
      float mn = fmaxf(M[i], mx);
      alpha[i] = __expf(M[i] - mn);
      M[i] = mn;
      float p0 = __expf(v0 - mn);
      float p1 = __expf(v1 - mn);
      L[i] = L[i] * alpha[i] + half_sum(p0 + p1);
      int pr = (i + 8 * hif) * 40;
      Pb[wave][pr + lh]      = (__bf16)p0;
      Pb[wave][pr + 16 + lh] = (__bf16)p1;
    }
#pragma unroll
    for (int n = 0; n < 4; ++n)
#pragma unroll
      for (int i = 0; i < 8; ++i)
        acc[n][i] *= alpha[i];

    // O += P(16x32) @ V(32x64) via 4 WMMAs (dh tiles of 16).
    v16bf pa = load_frag(&Pb[wave][lh * 40]);
#pragma unroll
    for (int n = 0; n < 4; ++n)
      acc[n] = wmma_bf16(pa, load_frag(&Vt[n * 16 + lh][0]), acc[n]);
    __syncthreads();
  }

  // Diagonal self term: s = (q . k_self)/8, always visible.
  {
    int srow = row0 + lh;
    const __bf16* qr = qb + (bhS + srow) * DH_ + hif * 32;
    const __bf16* kr = kb + (bhS + srow) * DH_ + hif * 32;
    float s = 0.0f;
#pragma unroll
    for (int j = 0; j < 32; ++j) s += (float)qr[j] * (float)kr[j];
    s += __shfl_xor(s, 16);
    Sself[wave][lh] = s * 0.125f;
  }
#pragma unroll
  for (int i = 0; i < 8; ++i) {
    int r = i + 8 * hif;
    float s  = Sself[wave][r];
    float mn = fmaxf(M[i], s);
    float a  = __expf(M[i] - mn);
    float p  = __expf(s - mn);
    M[i] = mn;
    L[i] = L[i] * a + p;
    int srow = row0 + r;
    const __bf16* vr = vb + (bhS + srow) * DH_;
#pragma unroll
    for (int n = 0; n < 4; ++n)
      acc[n][i] = acc[n][i] * a + p * (float)vr[n * 16 + lh];
  }

  // Normalize and store to (B, S, H*DH) bf16 (proj-GEMM friendly layout).
#pragma unroll
  for (int i = 0; i < 8; ++i) {
    float inv = 1.0f / L[i];
    int srow = row0 + i + 8 * hif;
    size_t base = ((size_t)b * S_ + srow) * E_ + h * DH_;
#pragma unroll
    for (int n = 0; n < 4; ++n)
      ob[base + n * 16 + lh] = (__bf16)(acc[n][i] * inv);
  }
}

// ---------------------------------------------------------------------------
// Kernel 3: output projection.  out[4096x1024] = A_bf16 @ W[1024x1024] + b
// A tile (128x32 bf16, 64B rows) staged via the Tensor Data Mover: the D#
// descriptor requests pad_amount=4 DWORDs every pad_interval=16 DWORDs, which
// reproduces the 40-element padded LDS row stride with zero VALU/VGPR cost.
// ---------------------------------------------------------------------------
__global__ __launch_bounds__(256) void proj_gemm_kernel(
    const unsigned short* __restrict__ A_, const float* __restrict__ W,
    const float* __restrict__ bias, float* __restrict__ out) {
  const __bf16* A = reinterpret_cast<const __bf16*>(A_);

  __shared__ __align__(16) __bf16 As[128][40];
  __shared__ __align__(16) __bf16 Bs[128][40];

  const int tid  = threadIdx.x;
  const int wave = tid >> 5;
  const int lane = tid & 31;
  const int lh   = lane & 15;
  const int hif  = lane >> 4;
  const int wm   = wave & 3;
  const int wn   = wave >> 2;
  const int gm0  = blockIdx.x * 128;
  const int gn0  = blockIdx.y * 128;

#if USE_TDM
  // LDS byte offset of As: low 32 bits of the generic address (LDS aperture
  // maps addr[31:0] to the LDS offset).
  const unsigned ldsAs = (unsigned)(size_t)(&As[0][0]);
#endif

  v8f acc[2][4] = {};

  for (int k0 = 0; k0 < E_; k0 += 32) {
#if USE_TDM
    if (tid < 32) {  // one wave drives the TDM; EXEC is ignored by tensor ops
      unsigned long long ga =
          (unsigned long long)(size_t)(A + (size_t)gm0 * E_ + k0);
      u32x4 g0;
      g0[0] = 1u;                                   // count=1, user descriptor
      g0[1] = ldsAs;                                // lds_addr (bytes)
      g0[2] = (unsigned)(ga & 0xffffffffu);         // global_addr[31:0]
      g0[3] = (unsigned)((ga >> 32) & 0x01ffffffu)  // global_addr[56:32]
              | (2u << 30);                         // type = 2 ("image")
      i32x8 g1;
      g1[0] = (int)((1u << 16)      // data_size = 2 bytes
                  | (1u << 20)      // pad_enable
                  | (3u << 22)      // pad_interval: 16 DWORDs (one 64B row)
                  | (3u << 25));    // pad_amount: 4 DWORDs (-> 80B row stride)
      g1[1] = (int)((E_ & 0xffff) << 16);           // tensor_dim0 lo16 @[63:48]
      g1[2] = (int)((E_ >> 16) | ((4096u & 0xffff) << 16)); // dim0 hi / dim1 lo
      g1[3] = (int)((4096u >> 16) | (32u << 16));   // dim1 hi / tile_dim0=32
      g1[4] = (int)(128u);                          // tile_dim1=128, tile_dim2=0
      g1[5] = (int)E_;                              // tensor_dim0_stride lo32
      g1[6] = 0;                                    // stride0 hi / stride1 lo
      g1[7] = 0;
      i32x4 gz = {0, 0, 0, 0};                      // 2D tensor: groups 2/3 unused
      i32x8 gz8 = {0, 0, 0, 0, 0, 0, 0, 0};         // extra group (6-arg form)
      __builtin_amdgcn_tensor_load_to_lds(g0, g1, gz, gz, gz8, 0);
      __builtin_amdgcn_s_wait_tensorcnt(0);
    }
#else
#pragma unroll
    for (int i = 0; i < 2; ++i) {            // manual A tile copy (bf16)
      int f = tid + i * 256;
      int r = f >> 2;
      int c = (f & 3) << 3;
      *(v8bf*)&As[r][c] = *(const v8bf*)(A + (size_t)(gm0 + r) * E_ + k0 + c);
    }
#endif
    // ---- load phase for W tile ----
    float4 wreg[4];
#pragma unroll
    for (int i = 0; i < 4; ++i) {
      int f  = tid + i * 256;
      int kr = f >> 5;
      int c  = (f & 31) << 2;
      wreg[i] = *(const float4*)(W + (size_t)(k0 + kr) * E_ + gn0 + c);
    }
    if (k0 + 32 < E_) {                      // next W tile prefetch into GL2
      int kr = tid >> 5, c2 = (tid & 31) << 2;
      __builtin_prefetch(W + (size_t)(k0 + 32 + kr) * E_ + gn0 + c2, 0, 3);
    }
    // ---- store phase: transpose to [n][k] bf16 ----
#pragma unroll
    for (int i = 0; i < 4; ++i) {
      int f  = tid + i * 256;
      int kr = f >> 5;
      int c  = (f & 31) << 2;
      Bs[c + 0][kr] = (__bf16)wreg[i].x;
      Bs[c + 1][kr] = (__bf16)wreg[i].y;
      Bs[c + 2][kr] = (__bf16)wreg[i].z;
      Bs[c + 3][kr] = (__bf16)wreg[i].w;
    }
    __syncthreads();

    v16bf af[2], bfg[4];
#pragma unroll
    for (int mi = 0; mi < 2; ++mi)
      af[mi] = load_frag(&As[wm * 32 + mi * 16 + lh][0]);
#pragma unroll
    for (int ni = 0; ni < 4; ++ni)
      bfg[ni] = load_frag(&Bs[wn * 64 + ni * 16 + lh][0]);
#pragma unroll
    for (int mi = 0; mi < 2; ++mi)
#pragma unroll
      for (int ni = 0; ni < 4; ++ni)
        acc[mi][ni] = wmma_bf16(af[mi], bfg[ni], acc[mi][ni]);
    __syncthreads();
  }

#pragma unroll
  for (int mi = 0; mi < 2; ++mi)
#pragma unroll
    for (int ni = 0; ni < 4; ++ni)
#pragma unroll
      for (int i = 0; i < 8; ++i) {
        int r = gm0 + wm * 32 + mi * 16 + i + 8 * hif;
        int c = gn0 + wn * 64 + ni * 16 + lh;
        out[(size_t)r * E_ + c] = acc[mi][ni][i] + bias[c];
      }
}

// ---------------------------------------------------------------------------
extern "C" void kernel_launch(void* const* d_in, const int* in_sizes, int n_in,
                              void* d_out, int out_size, void* d_ws, size_t ws_size,
                              hipStream_t stream) {
  (void)in_sizes; (void)n_in; (void)out_size; (void)ws_size;

  const float* hidden = (const float*)d_in[0];
  const float* pK     = (const float*)d_in[1];
  const float* pV     = (const float*)d_in[2];
  const float* tK     = (const float*)d_in[3];
  const float* tV     = (const float*)d_in[4];
  const unsigned char* pMask = (const unsigned char*)d_in[5];
  const float* attnW  = (const float*)d_in[6];
  const float* attnB  = (const float*)d_in[7];
  const float* projW  = (const float*)d_in[8];
  const float* projB  = (const float*)d_in[9];
  float* out = (float*)d_out;

  char* ws = (char*)d_ws;
  const size_t seg = (size_t)B_ * H_ * S_ * DH_ * sizeof(unsigned short); // 8 MiB
  unsigned short* qb = (unsigned short*)(ws);
  unsigned short* kb = (unsigned short*)(ws + seg);
  unsigned short* vb = (unsigned short*)(ws + 2 * seg);
  unsigned short* ob = (unsigned short*)(ws + 3 * seg);

  // 1) QKV projection: (4096x1024)@(1024x3072)
  qkv_gemm_kernel<<<dim3(32, 24), 256, 0, stream>>>(hidden, attnW, attnB,
                                                    qb, kb, vb);
  // 2) Attention: one block per (b,h, 64-row query block)
  attn_kernel<<<dim3(B_ * H_ * (S_ / 64)), 128, 0, stream>>>(
      pK, pV, tK, tV, pMask, qb, kb, vb, ob);
  // 3) Output projection: (4096x1024)@(1024x1024)
  proj_gemm_kernel<<<dim3(32, 8), 256, 0, stream>>>(ob, projW, projB, out);
}
